// cal_dy_90950227460803
// MI455X (gfx1250) — matrix operand order repaired and checked by reading, not compiled, not run on hardware
//
#include <hip/hip_runtime.h>
#include <hip/hip_bf16.h>

typedef __attribute__((ext_vector_type(16))) _Float16 v16h;
typedef __attribute__((ext_vector_type(8)))  _Float16 v8h;
typedef __attribute__((ext_vector_type(8)))  float    v8f;

#define CANVAS 512
#define CELLS  (512 * 512)
#define NB     2
#define NLI    1000000
#define NRA    500000

// ---------------------------------------------------------------------------
// Init: build symmetric shift matrix S (f16) and zero the dynamic maps D.
// S[a][b] = 1  iff |a-b| <= 2  or |a-b| == 511   (mod-513 wrap, phantom row
// 512 dropped -> only the two corner wrap entries survive).
// ---------------------------------------------------------------------------
__global__ void init_s_d_kernel(_Float16* __restrict__ S, _Float16* __restrict__ D) {
    int t = blockIdx.x * blockDim.x + threadIdx.x;
    if (t < CELLS) {
        int a = t >> 9, b = t & 511;
        int d = a - b; if (d < 0) d = -d;
        S[t] = (d <= 2 || d == 511) ? (_Float16)1.0f : (_Float16)0.0f;
    }
    if (t < NB * CELLS) D[t] = (_Float16)0.0f;
}

// ---------------------------------------------------------------------------
// Radar pass: emit ra_ind / ra_dy and scatter-mark dynamic voxels.
// (segment_max(|vr|) > 0.1  ==  any point with |vr| > 0.1 -> racy same-value
// stores are fine, no atomics needed.)
// ---------------------------------------------------------------------------
__global__ void ra_kernel(const float* __restrict__ ra_in,
                          const int*   __restrict__ ra_coords,
                          float* __restrict__ o_dy, float* __restrict__ o_ind,
                          void* __restrict__ dynmap, int f16map) {
    int t = blockIdx.x * blockDim.x + threadIdx.x;
    if (t >= NB * NRA) return;
    int2 yx = ((const int2*)ra_coords)[t];
    int flat = yx.x * CANVAS + yx.y;          // y * 512 + x
    float vr = fabsf(ra_in[5 * t + 4]);
    o_ind[t] = (float)flat;
    bool dyn = vr > 0.1f;
    o_dy[t] = dyn ? 1.0f : 0.0f;
    if (dyn) {
        int b = t / NRA;
        if (f16map) ((_Float16*)dynmap)[b * CELLS + flat] = (_Float16)1.0f;
        else        ((unsigned char*)dynmap)[b * CELLS + flat] = 1;
    }
}

// ---------------------------------------------------------------------------
// WMMA dilation GEMM:  Out^T = A x S   (S symmetric 512x512, f16).
// Called twice:  Tt = (D x S)^T,  M = (Tt x S)^T  ==  S * D * S  row-major.
// One wave per 16x16 tile, K swept 512 in 32-element WMMA steps.
//   A-frag  : lane l<16 -> row m_base+l, K chunks {0..7},{16..23} (+8 for hi)
//   B-frag  : lane l    -> col n_base+(l&15) = row of symmetric S,
//             K chunks {0..7},{16..23} (+8 for hi), same as A
//   D store : lane l -> Out[n_base+(l&15)][m_base+(l>>4)*8 .. +7]  (16B)
// EXEC is all-ones (no divergence), grid is an exact multiple.
// ---------------------------------------------------------------------------
__global__ __launch_bounds__(256) void gemm_dilate_kernel(
        const _Float16* __restrict__ A,
        const _Float16* __restrict__ S,
        _Float16* __restrict__ Out) {
    int wave  = (blockIdx.x * blockDim.x + threadIdx.x) >> 5;
    int lane  = threadIdx.x & 31;
    int batch = wave >> 10;                 // 1024 tiles per 512x512 matrix
    int tile  = wave & 1023;
    int m_base = (tile >> 5) << 4;
    int n_base = (tile & 31) << 4;
    int l16 = lane & 15;
    int hi  = lane >> 4;

    const _Float16* arow = A + (size_t)batch * CELLS + (size_t)(m_base + l16) * CANVAS;
    const _Float16* brow = S + (size_t)(n_base + l16) * CANVAS;

    v8f c = {};
#pragma unroll
    for (int k0 = 0; k0 < CANVAS; k0 += 32) {
        union { v16h v; v8h h[2]; } a;
        a.h[0] = *(const v8h*)(arow + k0 + hi * 8);        // K = k0+koff .. +7
        a.h[1] = *(const v8h*)(arow + k0 + hi * 8 + 16);   // K = k0+koff+16 .. +23
        union { v16h v; v8h h[2]; } bu;
        bu.h[0] = *(const v8h*)(brow + k0 + hi * 8);
        bu.h[1] = *(const v8h*)(brow + k0 + hi * 8 + 16);
        v16h b = bu.v;
        c = __builtin_amdgcn_wmma_f32_16x16x32_f16(
                false, a.v, false, b, (short)0, c, false, false);
    }
    v8h o;
#pragma unroll
    for (int j = 0; j < 8; ++j) o[j] = (_Float16)c[j];     // counts <= 25, exact
    *(v8h*)(Out + (size_t)batch * CELLS
                + (size_t)(n_base + l16) * CANVAS + m_base + hi * 8) = o;
}

// ---------------------------------------------------------------------------
// Lidar pass: emit li_ind and li_dy = dilated-map lookup.
// ---------------------------------------------------------------------------
__global__ void li_kernel(const int* __restrict__ li_coords,
                          float* __restrict__ o_dy, float* __restrict__ o_ind,
                          const void* __restrict__ neigh, int f16map) {
    int t = blockIdx.x * blockDim.x + threadIdx.x;
    if (t >= NB * NLI) return;
    int2 yx = ((const int2*)li_coords)[t];
    int flat = yx.x * CANVAS + yx.y;
    o_ind[t] = (float)flat;
    int b = t / NLI;
    bool dy;
    if (f16map) dy = (float)((const _Float16*)neigh)[b * CELLS + flat] > 0.5f;
    else        dy = ((const unsigned char*)neigh)[b * CELLS + flat] != 0;
    o_dy[t] = dy ? 1.0f : 0.0f;
}

// ---------------------------------------------------------------------------
// Fallback (tiny workspace): byte maps + 25-tap gather dilation.
// ---------------------------------------------------------------------------
__global__ void zero_bytes_kernel(unsigned char* p, int n) {
    int t = blockIdx.x * blockDim.x + threadIdx.x;
    if (t < n) p[t] = 0;
}

__global__ void dilate_fb_kernel(const unsigned char* __restrict__ dyn,
                                 unsigned char* __restrict__ neigh) {
    int t = blockIdx.x * blockDim.x + threadIdx.x;
    if (t >= NB * CELLS) return;
    int cell = t & (CELLS - 1);
    int ty = cell >> 9, tx = cell & 511;
    const unsigned char* d = dyn + (t - cell);
    int v = 0;
#pragma unroll
    for (int i = -2; i <= 2; ++i) {
        int sy = (ty + i + 513) % 513;
        if (sy >= 512) continue;
#pragma unroll
        for (int j = -2; j <= 2; ++j) {
            int sx = (tx + j + 513) % 513;
            if (sx >= 512) continue;
            v |= d[sy * 512 + sx];
        }
    }
    neigh[t] = (unsigned char)(v != 0);
}

// ---------------------------------------------------------------------------
extern "C" void kernel_launch(void* const* d_in, const int* in_sizes, int n_in,
                              void* d_out, int out_size, void* d_ws, size_t ws_size,
                              hipStream_t stream) {
    // inputs: 0=li_input(unused) 1=li_coords 2=li_idx(unused)
    //         3=ra_input 4=ra_coords 5=ra_idx(unused)
    const int*   li_coords = (const int*)d_in[1];
    const float* ra_in     = (const float*)d_in[3];
    const int*   ra_coords = (const int*)d_in[4];

    float* out      = (float*)d_out;                 // li_dy, li_ind, ra_dy, ra_ind
    float* o_li_dy  = out;
    float* o_li_ind = out + (size_t)NB * NLI;
    float* o_ra_dy  = out + (size_t)2 * NB * NLI;
    float* o_ra_ind = o_ra_dy + (size_t)NB * NRA;

    const int ra_blocks = (NB * NRA + 255) / 256;
    const int li_blocks = (NB * NLI + 255) / 256;

    size_t need_wmma = (size_t)(1 + 3 * NB) * CELLS * sizeof(_Float16); // 3.5 MB
    if (ws_size >= need_wmma) {
        _Float16* S  = (_Float16*)d_ws;
        _Float16* D  = S  + CELLS;            // NB maps
        _Float16* Tt = D  + (size_t)NB * CELLS;
        _Float16* M  = Tt + (size_t)NB * CELLS;

        init_s_d_kernel<<<(NB * CELLS + 255) / 256, 256, 0, stream>>>(S, D);
        ra_kernel<<<ra_blocks, 256, 0, stream>>>(ra_in, ra_coords,
                                                 o_ra_dy, o_ra_ind, D, 1);
        // 2048 waves (NB * 1024 tiles) per pass -> 256 blocks of 8 waves
        gemm_dilate_kernel<<<256, 256, 0, stream>>>(D,  S, Tt);  // Tt = (D*S)^T
        gemm_dilate_kernel<<<256, 256, 0, stream>>>(Tt, S, M);   // M  = S*D*S
        li_kernel<<<li_blocks, 256, 0, stream>>>(li_coords, o_li_dy, o_li_ind, M, 1);
    } else {
        unsigned char* dyn   = (unsigned char*)d_ws;
        unsigned char* neigh = dyn + (size_t)NB * CELLS;
        zero_bytes_kernel<<<(NB * CELLS + 255) / 256, 256, 0, stream>>>(dyn, NB * CELLS);
        ra_kernel<<<ra_blocks, 256, 0, stream>>>(ra_in, ra_coords,
                                                 o_ra_dy, o_ra_ind, dyn, 0);
        dilate_fb_kernel<<<(NB * CELLS + 255) / 256, 256, 0, stream>>>(dyn, neigh);
        li_kernel<<<li_blocks, 256, 0, stream>>>(li_coords, o_li_dy, o_li_ind, neigh, 0);
    }
}
